// GraphemeColourSynaesthesiaSpikeNet_35888746726081
// MI455X (gfx1250) — matrix operand order repaired and checked by reading, not compiled, not run on hardware
//
#include <hip/hip_runtime.h>

typedef __attribute__((ext_vector_type(16))) _Float16 v16h;
typedef __attribute__((ext_vector_type(8)))  float    v8f;
typedef __attribute__((ext_vector_type(4)))  float    v4f;
typedef __attribute__((ext_vector_type(4)))  unsigned u32x4;

#define M_DIM        512   // neurons (rows of K / W)
#define N_DIM        256   // flattened input dim; also replicated row count
#define ROWS_PER_WG  32
#define NUM_WG       16
#define KROW_PAD     520   // f16 elements per LDS row (512 + 8 pad -> bank-conflict avoidance)

__device__ __forceinline__ float sigmoidf(float x) {
  return 1.0f / (1.0f + __expf(-x));
}

__global__ void init_ws_kernel(unsigned* cnt) {
  if (threadIdx.x == 0) *cnt = 0u;
}

__global__ __launch_bounds__(256)
void spikenet_phase1(const float* __restrict__ x,
                     const float* __restrict__ W,
                     const float* __restrict__ K,
                     const int*   __restrict__ mi,
                     float*       __restrict__ out,
                     unsigned*    __restrict__ cnt,
                     float*       __restrict__ sraw /* 2 * 512 floats, double buffer */)
{
  __shared__ __align__(16) _Float16 Klds[ROWS_PER_WG * KROW_PAD]; // 33,280 B
  __shared__ __align__(16) float    ktmp[8 * M_DIM];              // 16 KB async staging
  __shared__ __align__(32) _Float16 s16[2 * M_DIM];   // [0,512): s as f16; [512,1024): zeros
  __shared__ float    sf[M_DIM];                      // current s as f32 (for I)
  __shared__ float    ylds[ROWS_PER_WG];              // K@s partial-sum accumulator
  __shared__ float    red[256];                       // max reduction

  const int tid  = threadIdx.x;
  const int g    = blockIdx.x;        // 0..15 : owns rows [g*32, g*32+32)
  const int lane = tid & 31;
  const int wid  = tid >> 5;          // 0..7
  const int col  = lane & 15;         // N-column within WMMA tile
  const int hi   = lane >> 4;         // 0/1 lane-half
  const int blk  = wid >> 2;          // 0/1 : which 16-row block of this WG
  const int kq   = wid & 3;           // k-quarter: 128 columns of K

  const int T = mi[0] * 100;          // max_iter / 0.01

  // s starts at zero; upper half of s16 is a permanent zero pad for B builds
  for (int idx = tid; idx < 2 * M_DIM; idx += 256) s16[idx] = (_Float16)0.0f;
  for (int idx = tid; idx < M_DIM; idx += 256)     sf[idx] = 0.0f;

  // ---- stage this WG's K slice into LDS via async global->LDS, convert to f16 ----
  {
    const unsigned tmpLds = (unsigned)(size_t)(&ktmp[0]); // flat low 32 bits == LDS byte addr
    for (int c = 0; c < 4; ++c) {           // 4 chunks of 8 rows (16 KB each)
      const char* gsrc =
          (const char*)(K + (size_t)(g * ROWS_PER_WG + c * 8) * M_DIM);
      #pragma unroll
      for (int i = 0; i < 4; ++i) {         // 256 lanes x 16 B = 4 KB per issue
        unsigned off = (unsigned)(i * 4096 + tid * 16);
        unsigned dst = tmpLds + off;
        unsigned long long src = (unsigned long long)(size_t)(gsrc + off);
        asm volatile("global_load_async_to_lds_b128 %0, %1, off"
                     :: "v"(dst), "v"(src) : "memory");
      }
      asm volatile("s_wait_asynccnt 0" ::: "memory");
      __syncthreads();
      for (int idx = tid; idx < 8 * M_DIM; idx += 256) {
        int r  = idx >> 9;                  // idx / 512
        int cc = idx & 511;
        Klds[(c * 8 + r) * KROW_PAD + cc] = (_Float16)ktmp[idx];
      }
      __syncthreads();                      // ktmp reusable next chunk
    }
  }

  // ---- per-owner constants/state (threads 0..31 each own one neuron m) ----
  float wxr = 0.0f;
  float vst = 0.1f;        // v0 = a = 0.1
  float ust = -12.25f;     // u0 = b*C = 0.2 * -61.25
  if (tid < ROWS_PER_WG) {
    int m = g * ROWS_PER_WG + tid;
    float acc = 0.0f;
    for (int k = 0; k < N_DIM; ++k) acc += W[(size_t)m * N_DIM + k] * x[k];
    wxr = acc;                                   // (W @ x.flatten())[m], constant
  }
  __syncthreads();

  const float dtc = 0.01f;
  const float Cc  = -61.25f;   // -65 + 15*0.25
  const float Dc  = 0.5f;      // 2 - 6*0.25

  // B source base: col 0 lanes read live s chunk, all other lanes the zero pad
  const _Float16* bbase = &s16[((col != 0) ? M_DIM : 0) + (hi ? 8 : 0)];
  const int m_loc = blk * 16 + col;
  const _Float16* krow = &Klds[m_loc * KROW_PAD];

  for (int t = 0; t < T; ++t) {
    if (tid < ROWS_PER_WG) ylds[tid] = 0.0f;
    __syncthreads();

    // ---- y = K_slice @ s via WMMA f32 <- f16 x f16 ----
    // wave handles rows [blk*16, blk*16+16), k in [kq*128, kq*128+128)
    union { v16h v; unsigned u[8]; } A, Bm;
    v8f acc = {};
    for (int c = 0; c < 4; ++c) {
      const int kc = kq * 128 + c * 32;
      // A (16x32 f16), documented layout:
      //   vgpr v, half-pair at k = (v<4?0:16) + (hi?8:0) + (v&3)*2
      #pragma unroll
      for (int v = 0; v < 8; ++v) {
        int k0 = kc + ((v & 4) ? 16 : 0) + (hi ? 8 : 0) + ((v & 3) * 2);
        A.u[v] = *(const unsigned*)(&krow[k0]);
      }
      // B (32x16 f16): column 0 = s chunk, other columns zero (zero-pad region)
      *(u32x4*)&Bm.u[0] = *(const u32x4*)(bbase + kc);
      *(u32x4*)&Bm.u[4] = *(const u32x4*)(bbase + kc + 16);
      acc = __builtin_amdgcn_wmma_f32_16x16x32_f16(
          false, A.v, false, Bm.v, (short)0, acc, false, false);
    }
    // D column 0: lane0 vgpr v = y[blk*16+v], lane16 vgpr v = y[blk*16+8+v]
    if (col == 0) {
      #pragma unroll
      for (int v = 0; v < 8; ++v)
        atomicAdd(&ylds[blk * 16 + hi * 8 + v], acc[v]);
    }
    __syncthreads();

    // ---- raw s for owned rows -> global exchange buffer (parity double-buffer) ----
    float* buf = sraw + (size_t)(t & 1) * M_DIM;
    if (tid < ROWS_PER_WG) {
      float sr = sigmoidf(wxr + ylds[tid]);
      __hip_atomic_store(&buf[g * ROWS_PER_WG + tid], sr,
                         __ATOMIC_RELAXED, __HIP_MEMORY_SCOPE_AGENT);
    }
    __syncthreads();

    // ---- device-wide step barrier (16 co-resident WGs) ----
    if (tid == 0) {
      __hip_atomic_fetch_add(cnt, 1u, __ATOMIC_ACQ_REL, __HIP_MEMORY_SCOPE_AGENT);
      const unsigned target = (unsigned)(t + 1) * NUM_WG;
      while (__hip_atomic_load(cnt, __ATOMIC_ACQUIRE, __HIP_MEMORY_SCOPE_AGENT) < target)
        __builtin_amdgcn_s_sleep(1);
    }
    __syncthreads();

    // ---- gather full raw s, max-normalize, clip ----
    float a0 = __hip_atomic_load(&buf[2 * tid],     __ATOMIC_RELAXED, __HIP_MEMORY_SCOPE_AGENT);
    float a1 = __hip_atomic_load(&buf[2 * tid + 1], __ATOMIC_RELAXED, __HIP_MEMORY_SCOPE_AGENT);
    red[tid] = fmaxf(a0, a1);
    __syncthreads();
    for (int off = 128; off > 0; off >>= 1) {
      if (tid < off) red[tid] = fmaxf(red[tid], red[tid + off]);
      __syncthreads();
    }
    const float inv = 1.5f / red[0];
    float s0 = fminf(fmaxf(a0 * inv, 0.01f), 1.5f);
    float s1 = fminf(fmaxf(a1 * inv, 0.01f), 1.5f);
    __syncthreads();
    sf[2 * tid]      = s0;
    sf[2 * tid + 1]  = s1;
    s16[2 * tid]     = (_Float16)s0;
    s16[2 * tid + 1] = (_Float16)s1;
    __syncthreads();

    // ---- Izhikevich update for owned rows; emit spikes row n=0 ----
    if (tid < ROWS_PER_WG) {
      int m = g * ROWS_PER_WG + tid;
      float I = sf[m];
      bool  fired = (vst >= 30.0f);
      float vr = fired ? Cc : vst;
      float ur = fired ? (ust + Dc) : ust;
      float dv = 0.04f * vr * vr + 5.0f * vr + 140.0f - ur + I;
      float du = 0.1f * (0.2f * vr - ur);
      vst = fired ? vr : (vr + dv * dtc);
      ust = fired ? ur : (ur + du * dtc);
      out[(size_t)t * (N_DIM * M_DIM) + m] = vst;  // row n = 0
    }
    // next iteration's leading __syncthreads() protects s16 reuse
  }
}

// Replicate row n=0 of each timestep to rows 1..255 with nontemporal b128 stores.
__global__ __launch_bounds__(256)
void spikenet_phase2(const int* __restrict__ mi, float* __restrict__ out)
{
  const int T     = mi[0] * 100;
  const int tid   = threadIdx.x;
  const int q4    = tid & 127;      // float4 index within a 512-float row
  const int nOff  = tid >> 7;       // 0/1
  const int nBase = blockIdx.y * 64;

  for (int t = blockIdx.x; t < T; t += gridDim.x) {
    const size_t base = (size_t)t * (N_DIM * M_DIM);
    v4f row = *(const v4f*)(&out[base + q4 * 4]);   // n=0 row chunk (L2-hot)
    for (int n = nBase + nOff; n < nBase + 64; n += 2) {
      if (n == 0) continue;
      __builtin_nontemporal_store(
          row, (v4f*)(&out[base + (size_t)n * M_DIM + q4 * 4]));
    }
  }
}

extern "C" void kernel_launch(void* const* d_in, const int* in_sizes, int n_in,
                              void* d_out, int out_size, void* d_ws, size_t ws_size,
                              hipStream_t stream) {
  (void)in_sizes; (void)n_in; (void)out_size; (void)ws_size;

  const float* x  = (const float*)d_in[0];   // (2,128) flattened -> 256
  const float* W  = (const float*)d_in[1];   // (512,256)
  const float* K  = (const float*)d_in[2];   // (512,512)
  const int*   mi = (const int*)d_in[3];     // max_iter (scalar)
  float* out = (float*)d_out;

  unsigned* cnt  = (unsigned*)d_ws;                    // step-barrier counter
  float*    sraw = (float*)((char*)d_ws + 256);        // 2*512 f32 exchange

  init_ws_kernel<<<1, 64, 0, stream>>>(cnt);
  spikenet_phase1<<<NUM_WG, 256, 0, stream>>>(x, W, K, mi, out, cnt, sraw);
  dim3 g2(1000, 4);
  spikenet_phase2<<<g2, 256, 0, stream>>>(mi, out);
}
